// SoftDecisionTree_42846593745236
// MI455X (gfx1250) — matrix operand order repaired and checked by reading, not compiled, not run on hardware
//
#include <hip/hip_runtime.h>
#include <math.h>

// ---------------- CDNA5 WMMA types ----------------
typedef __attribute__((ext_vector_type(16))) __bf16 v16bf;
typedef __attribute__((ext_vector_type(8)))  float  v8f;
typedef __attribute__((ext_vector_type(8)))  unsigned short us8;

// ---- workspace layout (byte offsets, all 256-aligned) ----
// header floats: s1_sum[20]@0 s1_sq[20]@20 a1[20]@40 b1[20]@60
//                s2_sum[50]@80 s2_sq[50]@130 a2[50]@180 b2[50]@230
//                s3_sum[10]@280 s3_sq[10]@290 a3[10]@300 b3[10]@310 P_prev[10]@320
#define HDR 512
// pooled1: bf16, channel stride padded 144 -> 160 (pad [144..159] is zero)
static const size_t B_HDR    = 0;
static const size_t B_POOL1  = 2048;                                  // bf16 8192*20*160
static const size_t B_FLAT   = B_POOL1 + (size_t)8192*20*160*2;       // bf16 8192*800
static const size_t B_FCPRE  = B_FLAT  + (size_t)8192*800*2;          // f32  8192*10
static const size_t B_H      = B_FCPRE + (size_t)8192*10*4;           // f32  8192*10
static const size_t B_P      = B_H     + (size_t)8192*10*4;           // f32  8192*255
static const size_t B_MU     = B_P     + (size_t)8192*255*4;          // bf16 8192*256
static const size_t B_MUT    = B_MU    + (size_t)8192*256*2;          // f32  256*8192
static const size_t B_PINEW  = B_MUT   + (size_t)8192*256*4;          // f32  256*10
static const size_t WS_BYTES = B_PINEW + 10240;                       // ~83 MB

// ---------------- helpers ----------------
__device__ __forceinline__ unsigned short f2bf(float f) {   // cold paths only
  union { float f; unsigned u; } v; v.f = f;
  unsigned r = v.u + 0x7FFFu + ((v.u >> 16) & 1u);          // round-nearest-even
  return (unsigned short)(r >> 16);
}
__device__ __forceinline__ __bf16 bfh(unsigned short s) {
  union { unsigned short u; __bf16 h; } v; v.u = s; return v.h;
}
// two contiguous 16B loads -> one 16-element bf16 operand vector
__device__ __forceinline__ v16bf ld16(const unsigned short* p0,
                                      const unsigned short* p1) {
  union { us8 h[2]; v16bf v; } u;
  u.h[0] = *(const us8*)p0;
  u.h[1] = *(const us8*)p1;
  return u.v;
}
__device__ __forceinline__ float wave_sum(float v) {
#pragma unroll
  for (int m = 16; m >= 1; m >>= 1) v += __shfl_xor(v, m, 32);
  return v;
}

// ---------------- init ----------------
__global__ void sdt_init_k(float* ws) {
  if (threadIdx.x < HDR) ws[threadIdx.x] = 0.f;
}

// ---------------- conv1 (K=25, VALU is right for this) ----------------
__global__ void sdt_conv1_stats_k(const float* __restrict__ x,
                                  const float* __restrict__ w,
                                  const float* __restrict__ b, float* ws) {
  __shared__ float sW[500], sB[20], shS[20], shQ[20];
  int tid = threadIdx.x;
  for (int i = tid; i < 500; i += 256) sW[i] = w[i];
  if (tid < 20) { sB[tid] = b[tid]; shS[tid] = 0.f; shQ[tid] = 0.f; }
  __syncthreads();
  int idx = blockIdx.x * 256 + tid;                   // exact: 8192*576 threads
  int n = idx / 576, pos = idx % 576, oy = pos / 24, ox = pos % 24;
  float patch[25];
#pragma unroll
  for (int q = 0; q < 25; ++q)
    patch[q] = x[n * 784 + (oy + q / 5) * 28 + (ox + q % 5)];
  int lane = tid & 31;
  for (int c = 0; c < 20; ++c) {
    float v = sB[c];
#pragma unroll
    for (int q = 0; q < 25; ++q) v += patch[q] * sW[c * 25 + q];
    float s = wave_sum(v), sq = wave_sum(v * v);
    if (lane == 0) { atomicAdd(&shS[c], s); atomicAdd(&shQ[c], sq); }
  }
  __syncthreads();
  if (tid < 20) { atomicAdd(&ws[0 + tid], shS[tid]); atomicAdd(&ws[20 + tid], shQ[tid]); }
}

__global__ void sdt_bn_fin_k(float* ws, const float* __restrict__ g,
                             const float* __restrict__ be,
                             int sumOff, int sqOff, int aOff, int bOff,
                             int C, float invN) {
  int c = threadIdx.x;
  if (c < C) {
    float m = ws[sumOff + c] * invN;
    float v = ws[sqOff + c] * invN - m * m;
    float a = g[c] * rsqrtf(v + 1e-5f);
    ws[aOff + c] = a;
    ws[bOff + c] = be[c] - m * a;
  }
}

__global__ void sdt_conv1_pool_k(const float* __restrict__ x,
                                 const float* __restrict__ w,
                                 const float* __restrict__ b,
                                 const float* __restrict__ ws,
                                 unsigned short* __restrict__ pooled1b) {
  __shared__ float sW[500], sB[20], sA[20], sBt[20];
  int tid = threadIdx.x;
  for (int i = tid; i < 500; i += 256) sW[i] = w[i];
  if (tid < 20) { sB[tid] = b[tid]; sA[tid] = ws[40 + tid]; sBt[tid] = ws[60 + tid]; }
  __syncthreads();
  int idx = blockIdx.x * 256 + tid;                   // exact: 8192*144 threads
  int n = idx / 144, pos = idx % 144, py = pos / 12, px = pos % 12;
  float patch[36];
#pragma unroll
  for (int i = 0; i < 36; ++i)
    patch[i] = x[n * 784 + (2 * py + i / 6) * 28 + (2 * px + i % 6)];
  for (int c = 0; c < 20; ++c) {
    float m = -1e30f;
#pragma unroll
    for (int dy = 0; dy < 2; ++dy)
#pragma unroll
      for (int dx = 0; dx < 2; ++dx) {
        float v = sB[c];
#pragma unroll
        for (int ky = 0; ky < 5; ++ky)
#pragma unroll
          for (int kx = 0; kx < 5; ++kx)
            v += patch[(dy + ky) * 6 + (dx + kx)] * sW[c * 25 + ky * 5 + kx];
        v = fmaxf(sA[c] * v + sBt[c], 0.f);
        m = fmaxf(m, v);
      }
    size_t cb = (size_t)(n * 20 + c) * 160;
    pooled1b[cb + py * 12 + px] = f2bf(m);            // bf16 store
    if (pos < 16) pooled1b[cb + 144 + pos] = 0;       // zero the K-pad region
  }
}

// ---------------- conv2 implicit GEMM (WMMA bf16) ----------------
// K is padded per-channel: 20 chunks of 32 (taps 25..31 zero).  A-offsets
// within a chunk are tile-invariant -> 16 precomputed offset registers; the
// channel loop is 16 u16 gathers + 2 ds_b128 + v_wmma.  Out-of-window taps
// read the zeroed pad [144..159] of the channel block.  Lanes with nn>=50
// are CLAMPED to B row 49 (unconditional LDS loads); the garbage lands only
// in C columns 50..63 which are never read.
#define C2_TILES  131072     // 8192 imgs * 4 row-pairs * 4 n-tiles
#define C2_BLOCKS 2048
#define C2_STRIDE (C2_BLOCKS * 8)

__device__ __forceinline__ void conv2_stage(const float* __restrict__ w,
                                            unsigned short* sB) {
  // LDS layout: sB[n][kp] , kp = c*32 + q (q<25 valid tap, else 0)
  for (int i = threadIdx.x; i < 50 * 640; i += 256) {
    int nn = i / 640, kp = i % 640, c = kp >> 5, q = kp & 31;
    sB[i] = (q < 25) ? f2bf(w[nn * 500 + c * 25 + q]) : (unsigned short)0;
  }
  __syncthreads();
}

__device__ __forceinline__ v8f conv2_tile(const unsigned short* __restrict__ chan,
                                          const unsigned short* brow,
                                          const int* offs) {
  v8f acc = {};
  for (int c = 0; c < 20; ++c) {
    v16bf a;
#pragma unroll
    for (int e = 0; e < 16; ++e) a[e] = bfh(chan[offs[e]]);
    v16bf b = ld16(brow + c * 32, brow + c * 32 + 8);   // unconditional
    acc = __builtin_amdgcn_wmma_f32_16x16x32_bf16(false, a, false, b,
                                                  (short)0, acc, false, false);
    chan += 160;
  }
  return acc;
}

__global__ void sdt_conv2_stats_k(const unsigned short* __restrict__ p1b,
                                  const float* __restrict__ w,
                                  const float* __restrict__ b, float* ws) {
  __shared__ __align__(16) unsigned short sB[50 * 640];
  conv2_stage(w, sB);
  int wid  = __builtin_amdgcn_readfirstlane(blockIdx.x * 8 + (threadIdx.x >> 5));
  int lane = threadIdx.x & 31;
  int t = lane & 15, hi = lane >> 4;
  int nt = wid & 3;                                    // fixed (stride % 4 == 0)
  int nn = nt * 16 + t;
  int nrow = (nn < 50) ? nn : 49;                      // clamp: see note above
  const unsigned short* brow = sB + nrow * 640 + 16 * hi;
  int rel[16];
#pragma unroll
  for (int e = 0; e < 16; ++e) {                       // tile-invariant tap offsets
    int q = 16 * hi + e;
    rel[e] = (q < 25) ? ((q / 5) * 12 + (q % 5)) : -1; // -1 => pad slot
  }
  float bias = (nn < 50) ? b[nn] : 0.f;
  float s = 0.f, sq = 0.f;
  for (int tIdx = wid; tIdx < C2_TILES; tIdx += C2_STRIDE) {
    int mt = tIdx >> 2, n = mt >> 2, r = mt & 3;
    int oyox = (2 * r + (t >> 3)) * 12 + (t & 7);
    int offs[16];
#pragma unroll
    for (int e = 0; e < 16; ++e) offs[e] = (rel[e] >= 0) ? (oyox + rel[e]) : 144;
    v8f acc = conv2_tile(p1b + (size_t)n * 3200, brow, offs);
#pragma unroll
    for (int v = 0; v < 8; ++v) { float xv = acc[v] + bias; s += xv; sq += xv * xv; }
  }
  s  += __shfl_xor(s, 16, 32);
  sq += __shfl_xor(sq, 16, 32);
  if (hi == 0 && nn < 50) {
    atomicAdd(&ws[80 + nn], s);
    atomicAdd(&ws[130 + nn], sq);
  }
}

__global__ void sdt_conv2_pool_k(const unsigned short* __restrict__ p1b,
                                 const float* __restrict__ w,
                                 const float* __restrict__ b,
                                 const float* __restrict__ ws,
                                 unsigned short* __restrict__ flatb) {
  __shared__ __align__(16) unsigned short sB[50 * 640];
  conv2_stage(w, sB);
  int wid  = __builtin_amdgcn_readfirstlane(blockIdx.x * 8 + (threadIdx.x >> 5));
  int lane = threadIdx.x & 31;
  int t = lane & 15, hi = lane >> 4;
  int nt = wid & 3;
  int nn = nt * 16 + t;
  int nrow = (nn < 50) ? nn : 49;                      // clamp: see note above
  const unsigned short* brow = sB + nrow * 640 + 16 * hi;
  int rel[16];
#pragma unroll
  for (int e = 0; e < 16; ++e) {
    int q = 16 * hi + e;
    rel[e] = (q < 25) ? ((q / 5) * 12 + (q % 5)) : -1;
  }
  float bias = (nn < 50) ? b[nn] : 0.f;
  float a2 = (nn < 50) ? ws[180 + nn] : 0.f;
  float b2 = (nn < 50) ? ws[230 + nn] : 0.f;
  for (int tIdx = wid; tIdx < C2_TILES; tIdx += C2_STRIDE) {
    int mt = tIdx >> 2, n = mt >> 2, r = mt & 3;
    int oyox = (2 * r + (t >> 3)) * 12 + (t & 7);
    int offs[16];
#pragma unroll
    for (int e = 0; e < 16; ++e) offs[e] = (rel[e] >= 0) ? (oyox + rel[e]) : 144;
    v8f acc = conv2_tile(p1b + (size_t)n * 3200, brow, offs);
    float m[4];
#pragma unroll
    for (int j = 0; j < 4; ++j) {                       // horizontal 2:1 pool
      float x0 = fmaxf(a2 * (acc[2 * j]     + bias) + b2, 0.f);
      float x1 = fmaxf(a2 * (acc[2 * j + 1] + bias) + b2, 0.f);
      m[j] = fmaxf(x0, x1);
    }
#pragma unroll
    for (int j = 0; j < 4; ++j)                         // vertical: lanes L <-> L+16
      m[j] = fmaxf(m[j], __shfl_xor(m[j], 16, 32));
    if (hi == 0 && nn < 50) {
      size_t base = (size_t)n * 800 + nn * 16 + r * 4;
#pragma unroll
      for (int j = 0; j < 4; ++j) flatb[base + j] = f2bf(m[j]);   // bf16 store
    }
  }
}

// ---------------- FC 800 -> 10 (WMMA, bf16 A from memory) ----------------
__global__ void sdt_fc1_k(const unsigned short* __restrict__ flatb,
                          const float* __restrict__ w,
                          const float* __restrict__ b,
                          float* ws, float* __restrict__ fc_pre) {
  __shared__ __align__(16) unsigned short sB[16 * 800];   // [n][k], rows 10..15 zero
  for (int i = threadIdx.x; i < 16 * 800; i += 256) {
    int nn = i / 800, k = i % 800;
    sB[i] = (nn < 10) ? f2bf(w[nn * 800 + k]) : (unsigned short)0;
  }
  __syncthreads();
  int wid  = __builtin_amdgcn_readfirstlane(blockIdx.x * 8 + (threadIdx.x >> 5));
  int lane = threadIdx.x & 31;
  int t = lane & 15, hi = lane >> 4;
  const unsigned short* arow = flatb + (size_t)(wid * 16 + t) * 800 + 8 * hi;
  const unsigned short* brow = sB + t * 800 + 8 * hi;
  v8f acc = {};
  for (int kc = 0; kc < 25; ++kc) {
    v16bf a  = ld16(arow + kc * 32, arow + kc * 32 + 16); // 2x global b128
    v16bf bb = ld16(brow + kc * 32, brow + kc * 32 + 16); // 2x ds b128
    acc = __builtin_amdgcn_wmma_f32_16x16x32_bf16(false, a, false, bb,
                                                  (short)0, acc, false, false);
  }
  int k = t;
  float bias = (k < 10) ? b[k] : 0.f;
  float s = 0.f, sq = 0.f;
#pragma unroll
  for (int v = 0; v < 8; ++v) {
    float xv = acc[v] + bias;
    int n = wid * 16 + v + 8 * hi;
    if (k < 10) fc_pre[n * 10 + k] = xv;
    s += xv; sq += xv * xv;
  }
  s  += __shfl_xor(s, 16, 32);
  sq += __shfl_xor(sq, 16, 32);
  if (hi == 0 && k < 10) { atomicAdd(&ws[280 + k], s); atomicAdd(&ws[290 + k], sq); }
}

__global__ void sdt_h_k(const float* __restrict__ fc_pre,
                        const float* __restrict__ ws, float* __restrict__ h) {
  int i = blockIdx.x * 256 + threadIdx.x;               // exact 81920
  int k = i % 10;
  h[i] = fmaxf(ws[300 + k] * fc_pre[i] + ws[310 + k], 0.f);
}

// ---------------- gates + leaf products ----------------
__global__ void sdt_theta_k(const float* __restrict__ h,
                            const float* __restrict__ tw,
                            const float* __restrict__ tb, float* __restrict__ p) {
  int i = blockIdx.x * 256 + threadIdx.x;               // exact 8192*255
  int n = i / 255, j = i % 255;
  float acc = tb[j];
  const float* hr = h + n * 10;
  const float* wr = tw + j * 10;
#pragma unroll
  for (int d = 0; d < 10; ++d) acc += hr[d] * wr[d];
  p[i] = 1.f / (1.f + expf(-acc));
}

__global__ void sdt_mu_k(const float* __restrict__ p,
                         unsigned short* __restrict__ mu16,
                         float* __restrict__ muT) {
  int i = blockIdx.x * 256 + threadIdx.x;               // exact 8192*256
  int n = i >> 8, l = i & 255;
  const float* pr = p + n * 255;
  float m = 1.f; int cur = 0;                           // preorder tree walk
#pragma unroll
  for (int d = 1; d <= 8; ++d) {
    int dir = (l >> (8 - d)) & 1;
    float f = pr[cur];
    m *= dir ? f : (1.f - f);
    cur += dir ? (1 << (8 - d)) : 1;
  }
  mu16[i] = f2bf(m);
  muT[(size_t)l * 8192 + n] = m;
}

__global__ void sdt_pprev_k(const float* __restrict__ pi, float* ws) {
  __shared__ float sh[10];
  if (threadIdx.x < 10) sh[threadIdx.x] = 0.f;
  __syncthreads();
  const float* pr = pi + threadIdx.x * 10;              // 256 threads = leaves
  float rs = 0.f;
#pragma unroll
  for (int k = 0; k < 10; ++k) rs += pr[k];
  float inv = 1.f / rs;
#pragma unroll
  for (int k = 0; k < 10; ++k) atomicAdd(&sh[k], pr[k] * inv);
  __syncthreads();
  if (threadIdx.x < 10) ws[320 + threadIdx.x] = sh[threadIdx.x] / 256.f;
}

__global__ void sdt_spinew_k(const float* __restrict__ muT,
                             const int* __restrict__ y,
                             const float* __restrict__ pi,
                             const float* __restrict__ ws,
                             float* __restrict__ pi_new) {
  __shared__ float cls[10], pih[10], dinv[1];
  int l = blockIdx.x;
  if (threadIdx.x < 10) cls[threadIdx.x] = 0.f;
  __syncthreads();
  const float* mt = muT + (size_t)l * 8192;
  for (int i = threadIdx.x; i < 8192; i += 256) atomicAdd(&cls[y[i]], mt[i]);
  __syncthreads();
  if (threadIdx.x < 10) {
    const float* pr = pi + l * 10;
    float rs = 0.f;
#pragma unroll
    for (int k = 0; k < 10; ++k) rs += pr[k];
    float pin = pr[threadIdx.x] / rs;
    pih[threadIdx.x] = pin * (cls[threadIdx.x] / ws[320 + threadIdx.x]);
  }
  __syncthreads();
  if (threadIdx.x == 0) {
    float d = 0.f;
#pragma unroll
    for (int k = 0; k < 10; ++k) d += pih[k];
    dinv[0] = 1.f / d;
  }
  __syncthreads();
  if (threadIdx.x < 10) pi_new[l * 10 + threadIdx.x] = pih[threadIdx.x] * dinv[0];
}

// ---------------- P = mu @ pi_new ; out = log(P)  (WMMA) ----------------
__global__ void sdt_pfinal_k(const unsigned short* __restrict__ mu16,
                             const float* __restrict__ pi_new,
                             float* __restrict__ out) {
  __shared__ __align__(16) unsigned short sB[16 * 256];   // [n][k], rows 10..15 zero
  for (int i = threadIdx.x; i < 16 * 256; i += 256) {
    int nn = i >> 8, k = i & 255;
    sB[i] = (nn < 10) ? f2bf(pi_new[k * 10 + nn]) : (unsigned short)0;
  }
  __syncthreads();
  int wid  = __builtin_amdgcn_readfirstlane(blockIdx.x * 8 + (threadIdx.x >> 5));
  int lane = threadIdx.x & 31;
  int t = lane & 15, hi = lane >> 4;
  const unsigned short* arow = mu16 + (size_t)(wid * 16 + t) * 256 + 8 * hi;
  const unsigned short* brow = sB + t * 256 + 8 * hi;
  v8f acc = {};
#pragma unroll
  for (int kc = 0; kc < 8; ++kc) {
    v16bf a  = ld16(arow + kc * 32, arow + kc * 32 + 16);
    v16bf bb = ld16(brow + kc * 32, brow + kc * 32 + 16);
    acc = __builtin_amdgcn_wmma_f32_16x16x32_bf16(false, a, false, bb,
                                                  (short)0, acc, false, false);
  }
  if (t < 10) {
#pragma unroll
    for (int v = 0; v < 8; ++v) {
      int n = wid * 16 + v + 8 * hi;
      out[n * 10 + t] = logf(acc[v]);
    }
  }
}

// ---------------- launch ----------------
extern "C" void kernel_launch(void* const* d_in, const int* in_sizes, int n_in,
                              void* d_out, int out_size, void* d_ws, size_t ws_size,
                              hipStream_t stream) {
  (void)in_sizes; (void)n_in; (void)out_size;
  const float* x       = (const float*)d_in[0];
  const int*   y       = (const int*)  d_in[1];
  const float* conv1_w = (const float*)d_in[2];
  const float* conv1_b = (const float*)d_in[3];
  const float* conv2_w = (const float*)d_in[4];
  const float* conv2_b = (const float*)d_in[5];
  const float* g1      = (const float*)d_in[6];
  const float* be1     = (const float*)d_in[7];
  const float* g2      = (const float*)d_in[8];
  const float* be2     = (const float*)d_in[9];
  const float* fc1_w   = (const float*)d_in[10];
  const float* fc1_b   = (const float*)d_in[11];
  const float* g3      = (const float*)d_in[12];
  const float* be3     = (const float*)d_in[13];
  const float* theta_w = (const float*)d_in[14];
  const float* theta_b = (const float*)d_in[15];
  const float* pi      = (const float*)d_in[16];
  float* out = (float*)d_out;

  if (ws_size < WS_BYTES) return;                       // ~83 MB scratch needed
  char* wsb = (char*)d_ws;
  float*          ws      = (float*)(wsb + B_HDR);
  unsigned short* pooled1 = (unsigned short*)(wsb + B_POOL1);
  unsigned short* flatb   = (unsigned short*)(wsb + B_FLAT);
  float*          fc_pre  = (float*)(wsb + B_FCPRE);
  float*          h       = (float*)(wsb + B_H);
  float*          p       = (float*)(wsb + B_P);
  unsigned short* mu16    = (unsigned short*)(wsb + B_MU);
  float*          muT     = (float*)(wsb + B_MUT);
  float*          pi_new  = (float*)(wsb + B_PINEW);

  sdt_init_k<<<1, HDR, 0, stream>>>(ws);

  // conv1: stats -> finalize -> fused BN/ReLU/pool (bf16 out, K-pad zeroed)
  sdt_conv1_stats_k<<<18432, 256, 0, stream>>>(x, conv1_w, conv1_b, ws);
  sdt_bn_fin_k<<<1, 64, 0, stream>>>(ws, g1, be1, 0, 20, 40, 60, 20,
                                     1.f / (8192.f * 576.f));
  sdt_conv1_pool_k<<<4608, 256, 0, stream>>>(x, conv1_w, conv1_b, ws, pooled1);

  // conv2 (WMMA implicit GEMM): stats -> finalize -> fused BN/ReLU/pool
  sdt_conv2_stats_k<<<C2_BLOCKS, 256, 0, stream>>>(pooled1, conv2_w, conv2_b, ws);
  sdt_bn_fin_k<<<1, 64, 0, stream>>>(ws, g2, be2, 80, 130, 180, 230, 50,
                                     1.f / (8192.f * 64.f));
  sdt_conv2_pool_k<<<C2_BLOCKS, 256, 0, stream>>>(pooled1, conv2_w, conv2_b, ws, flatb);

  // fc1 (WMMA) + BN stats -> finalize -> relu
  sdt_fc1_k<<<64, 256, 0, stream>>>(flatb, fc1_w, fc1_b, ws, fc_pre);
  sdt_bn_fin_k<<<1, 64, 0, stream>>>(ws, g3, be3, 280, 290, 300, 310, 10,
                                     1.f / 8192.f);
  sdt_h_k<<<320, 256, 0, stream>>>(fc_pre, ws, h);

  // gates, leaf products, leaf statistics
  sdt_theta_k<<<8160, 256, 0, stream>>>(h, theta_w, theta_b, p);
  sdt_mu_k<<<8192, 256, 0, stream>>>(p, mu16, muT);
  sdt_pprev_k<<<1, 256, 0, stream>>>(pi, ws);
  sdt_spinew_k<<<256, 256, 0, stream>>>(muT, y, pi, ws, pi_new);

  // final P = mu @ pi_new (WMMA), log epilogue
  sdt_pfinal_k<<<64, 256, 0, stream>>>(mu16, pi_new, out);
}
